// MultiHeadedAttention_44521630991052
// MI455X (gfx1250) — matrix operand order repaired and checked by reading, not compiled, not run on hardware
//
#include <hip/hip_runtime.h>
#include <hip/hip_bf16.h>

#define B_  4
#define S_  2048
#define D_  512
#define H_  8
#define DK_ 64

typedef __attribute__((ext_vector_type(16))) _Float16 v16h;
typedef __attribute__((ext_vector_type(8)))  _Float16 v8h;
typedef __attribute__((ext_vector_type(8)))  float    v8f;
typedef __attribute__((ext_vector_type(4)))  float    v4f;

#define WMMA_F16(a, b, c) \
    __builtin_amdgcn_wmma_f32_16x16x32_f16(false, (a), false, (b), (short)0, (c), false, false)

// ---------------- fragment loaders (layouts per CDNA5 ISA 7.12.2) ----------------

// A fragment (16x32, f16 source, row-major, lda in elems). base already at (row0, k0).
// lane L: row = L&15 ; halves 0..7 = K[(L>>4)*8 .. +7], halves 8..15 = K[(L>>4)*8+16 .. +7]
__device__ __forceinline__ v16h load_a_f16(const _Float16* __restrict__ base, int lda, int lane) {
    const _Float16* p = base + (lane & 15) * lda + ((lane >> 4) << 3);
    v8h lo = *(const v8h*)(p);
    v8h hi = *(const v8h*)(p + 16);
    v16h a;
#pragma unroll
    for (int i = 0; i < 8; ++i) { a[i] = lo[i]; a[8 + i] = hi[i]; }
    return a;
}

// A fragment from fp32 source with on-the-fly f32->f16 convert
__device__ __forceinline__ v16h load_a_f32cvt(const float* __restrict__ base, int lda, int lane) {
    const float* p = base + (lane & 15) * lda + ((lane >> 4) << 3);
    v4f a0 = *(const v4f*)(p);
    v4f a1 = *(const v4f*)(p + 4);
    v4f b0 = *(const v4f*)(p + 16);
    v4f b1 = *(const v4f*)(p + 20);
    v16h a;
#pragma unroll
    for (int i = 0; i < 4; ++i) {
        a[i]      = (_Float16)a0[i];
        a[4 + i]  = (_Float16)a1[i];
        a[8 + i]  = (_Float16)b0[i];
        a[12 + i] = (_Float16)b1[i];
    }
    return a;
}

// B fragment (32x16) from transposed storage Bt[N, Ktot] row-major. base at (n0, k0).
// lane L: col n = L&15 ; halves 0..7 = K[(L>>4)*8 .. +7], halves 8..15 = K[(L>>4)*8+16 .. +7]
__device__ __forceinline__ v16h load_b_f16(const _Float16* __restrict__ base, int ldb, int lane) {
    const _Float16* p = base + (lane & 15) * ldb + ((lane >> 4) << 3);
    v8h lo = *(const v8h*)(p);
    v8h hi = *(const v8h*)(p + 16);
    v16h b;
#pragma unroll
    for (int i = 0; i < 8; ++i) { b[i] = lo[i]; b[8 + i] = hi[i]; }
    return b;
}

// ---------------- kernel 0: weight transpose + f32->f16 (512x512) ----------------
__global__ __launch_bounds__(256) void wtrans_kernel(const float* __restrict__ W,
                                                     _Float16* __restrict__ WT) {
    int idx = blockIdx.x * 256 + threadIdx.x;   // 0..262143
    int n = idx >> 9, k = idx & 511;
    WT[(size_t)n * 512 + k] = (_Float16)W[(size_t)k * 512 + n];
}

// ---------------- kernel 1: QKV projection GEMM ----------------
// X[B*S, 512] (f32) @ W[512, 512] -> head-major f16 output.
// mode 0/1: dst[((b*H+h)*S + s)*64 + d]   (Q, K)
// mode 2:   dst[((b*H+h)*64 + d)*S + s]   (V transposed)
__global__ __launch_bounds__(256) void proj_kernel(const float* __restrict__ X,
                                                   const _Float16* __restrict__ WT,
                                                   _Float16* __restrict__ dst, int mode) {
    int tid = threadIdx.x, lane = tid & 31, w = tid >> 5;
    int rowBase = blockIdx.y * 64 + ((w & 3) << 4);
    int colBase = blockIdx.x * 64 + ((w >> 2) << 5);
    v8f acc0 = {}, acc1 = {};
    const float* Ab = X + (size_t)rowBase * 512;
    const _Float16* Bb0 = WT + (size_t)colBase * 512;
    const _Float16* Bb1 = WT + (size_t)(colBase + 16) * 512;
#pragma unroll 4
    for (int k0 = 0; k0 < 512; k0 += 32) {
        v16h a  = load_a_f32cvt(Ab + k0, 512, lane);
        v16h b0 = load_b_f16(Bb0 + k0, 512, lane);
        acc0 = WMMA_F16(a, b0, acc0);
        v16h b1 = load_b_f16(Bb1 + k0, 512, lane);
        acc1 = WMMA_F16(a, b1, acc1);
    }
    int r0 = rowBase + ((lane >> 4) << 3);
#pragma unroll
    for (int j = 0; j < 2; ++j) {
        int col = colBase + j * 16 + (lane & 15);
        int hh = col >> 6, d = col & 63;
#pragma unroll
        for (int r = 0; r < 8; ++r) {
            int t = r0 + r;
            int b = t >> 11, s = t & (S_ - 1);
            float v = (j == 0) ? acc0[r] : acc1[r];
            if (mode == 2)
                dst[(((size_t)(b * H_ + hh)) * 64 + d) * S_ + s] = (_Float16)v;
            else
                dst[(((size_t)(b * H_ + hh)) * S_ + s) * 64 + d] = (_Float16)v;
        }
    }
}

// ---------------- kernel 2: raw scores = Q @ K^T * (1/sqrt(DK)) ----------------
__global__ __launch_bounds__(256) void scores_kernel(const _Float16* __restrict__ Q16,
                                                     const _Float16* __restrict__ K16,
                                                     float* __restrict__ scores) {
    int tid = threadIdx.x, lane = tid & 31, w = tid >> 5;
    int bh = blockIdx.z;
    int qBase = blockIdx.y * 64 + ((w & 3) << 4);
    int kBase = blockIdx.x * 128 + ((w >> 2) << 6);
    const _Float16* Qb = Q16 + ((size_t)bh * S_ + qBase) * 64;
    const _Float16* Kb = K16 + (size_t)bh * S_ * 64;
    v8f acc[4] = {{}, {}, {}, {}};
#pragma unroll
    for (int k0 = 0; k0 < 64; k0 += 32) {
        v16h a = load_a_f16(Qb + k0, 64, lane);
#pragma unroll
        for (int j = 0; j < 4; ++j) {
            v16h b = load_b_f16(Kb + (size_t)(kBase + j * 16) * 64 + k0, 64, lane);
            acc[j] = WMMA_F16(a, b, acc[j]);
        }
    }
    int r0 = qBase + ((lane >> 4) << 3);
#pragma unroll
    for (int j = 0; j < 4; ++j) {
        int col = kBase + j * 16 + (lane & 15);
#pragma unroll
        for (int r = 0; r < 8; ++r)
            scores[((size_t)bh * S_ + (r0 + r)) * S_ + col] = acc[j][r] * 0.125f;
    }
}

// ---------------- kernel 3: softmax in place (one block per row of 2048) -------
__global__ __launch_bounds__(256) void softmax_kernel(float* __restrict__ scores) {
    float* p = scores + (size_t)blockIdx.x * S_;
    int tid = threadIdx.x;
    __shared__ float red[256];
    float v[8];
    float m = -1e30f;
#pragma unroll
    for (int j = 0; j < 8; ++j) { v[j] = p[tid + j * 256]; m = fmaxf(m, v[j]); }
    red[tid] = m; __syncthreads();
    for (int s = 128; s > 0; s >>= 1) {
        if (tid < s) red[tid] = fmaxf(red[tid], red[tid + s]);
        __syncthreads();
    }
    m = red[0]; __syncthreads();
    float sum = 0.f;
#pragma unroll
    for (int j = 0; j < 8; ++j) { v[j] = __expf(v[j] - m); sum += v[j]; }
    red[tid] = sum; __syncthreads();
    for (int s = 128; s > 0; s >>= 1) {
        if (tid < s) red[tid] += red[tid + s];
        __syncthreads();
    }
    float inv = 1.0f / red[0];
#pragma unroll
    for (int j = 0; j < 8; ++j) p[tid + j * 256] = v[j] * inv;
}

// ---------------- kernel 4: context = P @ V ----------------
// A = probs (f32 -> f16 on the fly), B = V^T [DV, S] f16. ctx f16 [B*S, H*DV]
__global__ __launch_bounds__(256) void context_kernel(const float* __restrict__ probs,
                                                      const _Float16* __restrict__ Vt,
                                                      _Float16* __restrict__ ctx) {
    int tid = threadIdx.x, lane = tid & 31, w = tid >> 5;
    int bh = blockIdx.y;
    int qBase = blockIdx.x * 64 + ((w & 3) << 4);
    int dvBase = (w >> 2) << 5;
    const float* Pb = probs + ((size_t)bh * S_ + qBase) * S_;
    const _Float16* Vb = Vt + (size_t)bh * 64 * S_;
    v8f acc[2] = {{}, {}};
    for (int k0 = 0; k0 < S_; k0 += 32) {
        __builtin_prefetch(Pb + (lane & 15) * S_ + k0 + 64, 0, 0);
        v16h a = load_a_f32cvt(Pb + k0, S_, lane);
#pragma unroll
        for (int j = 0; j < 2; ++j) {
            v16h b = load_b_f16(Vb + (size_t)(dvBase + j * 16) * S_ + k0, S_, lane);
            acc[j] = WMMA_F16(a, b, acc[j]);
        }
    }
    int b = bh >> 3, hh = bh & 7;
    int r0 = qBase + ((lane >> 4) << 3);
#pragma unroll
    for (int j = 0; j < 2; ++j) {
        int col = hh * 64 + dvBase + j * 16 + (lane & 15);
#pragma unroll
        for (int r = 0; r < 8; ++r)
            ctx[(size_t)(b * S_ + r0 + r) * 512 + col] = (_Float16)acc[j][r];
    }
}

// ---------------- kernel 5: out proj + residual + LayerNorm ----------------
__global__ __launch_bounds__(256) void outln_kernel(const _Float16* __restrict__ ctx,
                                                    const _Float16* __restrict__ WoT,
                                                    const float* __restrict__ resid,
                                                    const float* __restrict__ gamma,
                                                    const float* __restrict__ beta,
                                                    float* __restrict__ y) {
    __shared__ float xs[16][513];
    __shared__ float ps[16][16], pq[16][16];
    __shared__ float smu[16], sinv[16];
    int tid = threadIdx.x, lane = tid & 31, w = tid >> 5;
    int rowBase = blockIdx.x * 16;
    int nBase = w * 64;
    v8f acc[4] = {{}, {}, {}, {}};
    const _Float16* Ab = ctx + (size_t)rowBase * 512;
#pragma unroll 4
    for (int k0 = 0; k0 < 512; k0 += 32) {
        v16h a = load_a_f16(Ab + k0, 512, lane);
#pragma unroll
        for (int j = 0; j < 4; ++j) {
            v16h b = load_b_f16(WoT + (size_t)(nBase + j * 16) * 512 + k0, 512, lane);
            acc[j] = WMMA_F16(a, b, acc[j]);
        }
    }
    int r0 = (lane >> 4) << 3;
#pragma unroll
    for (int j = 0; j < 4; ++j) {
        int col = nBase + j * 16 + (lane & 15);
#pragma unroll
        for (int r = 0; r < 8; ++r) {
            int row = r0 + r;
            xs[row][col] = acc[j][r] + resid[(size_t)(rowBase + row) * 512 + col];
        }
    }
    __syncthreads();
    int row = tid >> 4, seg = tid & 15;
    float s1 = 0.f, s2 = 0.f;
#pragma unroll 8
    for (int c = seg * 32; c < seg * 32 + 32; ++c) {
        float x = xs[row][c];
        s1 += x; s2 += x * x;
    }
    ps[row][seg] = s1; pq[row][seg] = s2;
    __syncthreads();
    if (seg == 0) {
        float a = 0.f, b = 0.f;
#pragma unroll
        for (int i = 0; i < 16; ++i) { a += ps[row][i]; b += pq[row][i]; }
        float mu = a * (1.0f / 512.0f);
        float var = b * (1.0f / 512.0f) - mu * mu;
        smu[row] = mu;
        sinv[row] = rsqrtf(var + 1e-5f);
    }
    __syncthreads();
    float mu = smu[row], inv = sinv[row];
#pragma unroll 8
    for (int c = seg * 32; c < seg * 32 + 32; ++c)
        y[(size_t)(rowBase + row) * 512 + c] = (xs[row][c] - mu) * inv * gamma[c] + beta[c];
}

// ---------------- host side ----------------
extern "C" void kernel_launch(void* const* d_in, const int* in_sizes, int n_in,
                              void* d_out, int out_size, void* d_ws, size_t ws_size,
                              hipStream_t stream) {
    const float* inQ   = (const float*)d_in[0];
    const float* inK   = (const float*)d_in[1];
    const float* inV   = (const float*)d_in[2];
    const float* WQ    = (const float*)d_in[3];
    const float* WK    = (const float*)d_in[4];
    const float* WV    = (const float*)d_in[5];
    const float* WO    = (const float*)d_in[6];
    const float* gamma = (const float*)d_in[7];
    const float* beta  = (const float*)d_in[8];

    float* y      = (float*)d_out;
    float* scores = y + (size_t)B_ * S_ * D_;     // tuple output: (y, scores)

    // workspace layout (f16 elements)
    _Float16* ws = (_Float16*)d_ws;
    const size_t WSZ = 512 * 512;                 // one transposed weight
    const size_t ACT = (size_t)B_ * H_ * S_ * 64; // one head-major activation
    _Float16* WqT = ws;
    _Float16* WkT = WqT + WSZ;
    _Float16* WvT = WkT + WSZ;
    _Float16* WoT = WvT + WSZ;
    _Float16* Q16 = WoT + WSZ;
    _Float16* K16 = Q16 + ACT;
    _Float16* Vt  = K16 + ACT;
    _Float16* ctx = Vt  + ACT;

    // 0) weights -> f16 transposed
    wtrans_kernel<<<1024, 256, 0, stream>>>(WQ, WqT);
    wtrans_kernel<<<1024, 256, 0, stream>>>(WK, WkT);
    wtrans_kernel<<<1024, 256, 0, stream>>>(WV, WvT);
    wtrans_kernel<<<1024, 256, 0, stream>>>(WO, WoT);

    // 1) QKV projections
    dim3 pgrid(8, (B_ * S_) / 64);
    proj_kernel<<<pgrid, 256, 0, stream>>>(inQ, WqT, Q16, 0);
    proj_kernel<<<pgrid, 256, 0, stream>>>(inK, WkT, K16, 1);
    proj_kernel<<<pgrid, 256, 0, stream>>>(inV, WvT, Vt, 2);

    // 2) raw scaled scores into d_out
    dim3 sgrid(S_ / 128, S_ / 64, B_ * H_);
    scores_kernel<<<sgrid, 256, 0, stream>>>(Q16, K16, scores);

    // 3) softmax in place (these are the returned probabilities)
    softmax_kernel<<<B_ * H_ * S_, 256, 0, stream>>>(scores);

    // 4) context = P @ V
    dim3 cgrid(S_ / 64, B_ * H_);
    context_kernel<<<cgrid, 256, 0, stream>>>(scores, Vt, ctx);

    // 5) output projection + residual + LayerNorm
    outln_kernel<<<(B_ * S_) / 16, 256, 0, stream>>>(ctx, WoT, inQ, gamma, beta, y);
}
